// Model_81071802680076
// MI455X (gfx1250) — matrix operand order repaired and checked by reading, not compiled, not run on hardware
//
#include <hip/hip_runtime.h>
#include <math.h>

// ---------------- types ----------------
typedef __attribute__((ext_vector_type(16))) __bf16 v16bf;
typedef __attribute__((ext_vector_type(8)))  __bf16 bf16x8;
typedef __attribute__((ext_vector_type(8)))  float  v8f;

static __device__ inline v8f wmma_bf16(v16bf a, v16bf b, v8f c) {
  // D = A(16x32) * B(32x16) + C(16x16), f32 accumulate
  return __builtin_amdgcn_wmma_f32_16x16x32_bf16(false, a, false, b, (short)0, c, false, false);
}
// A-fragment: elements 0..7 = K(8g..8g+7), 8..15 = K(16+8g..): lo at p, hi at p+16
static __device__ inline v16bf ld_frag_a(const __bf16* p) {
  bf16x8 lo = *(const bf16x8*)p;
  bf16x8 hi = *(const bf16x8*)(p + 16);
  return __builtin_shufflevector(lo, hi, 0,1,2,3,4,5,6,7,8,9,10,11,12,13,14,15);
}
// B-fragment: elements 0..7 = K(8g..8g+7), 8..15 = K(16+8g..): lo at p, hi at p+16
static __device__ inline v16bf ld_frag_b(const __bf16* p) {
  bf16x8 lo = *(const bf16x8*)p;
  bf16x8 hi = *(const bf16x8*)(p + 16);
  return __builtin_shufflevector(lo, hi, 0,1,2,3,4,5,6,7,8,9,10,11,12,13,14,15);
}
static __device__ inline bf16x8 cvt8(float4 v0, float4 v1) {
  bf16x8 w;
  w[0] = (__bf16)v0.x; w[1] = (__bf16)v0.y; w[2] = (__bf16)v0.z; w[3] = (__bf16)v0.w;
  w[4] = (__bf16)v1.x; w[5] = (__bf16)v1.y; w[6] = (__bf16)v1.z; w[7] = (__bf16)v1.w;
  return w;
}

// ---------------- constants ----------------
#define DLLM   4096
#define NTOK   1000
#define VSZ    32000
#define HF     256          // N_HEADS * D_FF
#define NKV    512          // k + v projections concatenated
#define BN_CNT 224          // 32 * 7
#define LPAT   64           // patch count
#define DMODEL 32

// =====================================================================
// Kernel 1: convert wk|wv rows to bf16 ([n][k] layout == WMMA-B friendly),
// plus row sums (to fold map_b through the reassociated GEMM) and biases.
// =====================================================================
__global__ __launch_bounds__(256)
void k_prep_bkv(const float* __restrict__ wk, const float* __restrict__ wv,
                const float* __restrict__ bk, const float* __restrict__ bv,
                __bf16* __restrict__ Bkv, float* __restrict__ skv, float* __restrict__ bkvb) {
  __shared__ float red[256];
  const int h = blockIdx.x;                 // 0..511
  const int tid = threadIdx.x;
  const float* src = (h < HF) ? (wk + (size_t)h * DLLM) : (wv + (size_t)(h - HF) * DLLM);
  float s = 0.f;
  for (int j = tid; j < DLLM; j += 256) {
    float v = src[j];
    Bkv[(size_t)h * DLLM + j] = (__bf16)v;
    s += v;
  }
  red[tid] = s; __syncthreads();
  for (int o = 128; o > 0; o >>= 1) { if (tid < o) red[tid] += red[tid + o]; __syncthreads(); }
  if (tid == 0) {
    skv[h]  = red[0];
    bkvb[h] = (h < HF) ? bk[h] : bv[h - HF];
  }
}

// =====================================================================
// Kernel 2: per-series stats + normalization + patching + circular conv.
// One block per (b,n) series. enc[bn][l][o], l<64, o<32.
// =====================================================================
__global__ __launch_bounds__(256)
void k_stats_enc(const float* __restrict__ x_enc, const float* __restrict__ conv_w,
                 float* __restrict__ enc, float* __restrict__ meanw, float* __restrict__ stdw) {
  __shared__ float xn[528];
  __shared__ float red[256];
  __shared__ float cw2[1536];     // [ (i*3+t)*32 + o ]
  __shared__ float stats[2];
  const int bn = blockIdx.x, b = bn / 7, n = bn - b * 7;
  const int tid = threadIdx.x;

  float s = 0.f, sq = 0.f;
  for (int t = tid; t < 512; t += 256) {
    float v = x_enc[((size_t)b * 512 + t) * 7 + n];
    xn[t] = v; s += v; sq += v * v;
  }
  red[tid] = s; __syncthreads();
  for (int o = 128; o > 0; o >>= 1) { if (tid < o) red[tid] += red[tid + o]; __syncthreads(); }
  if (tid == 0) stats[0] = red[0] * (1.f / 512.f);
  __syncthreads();
  red[tid] = sq; __syncthreads();
  for (int o = 128; o > 0; o >>= 1) { if (tid < o) red[tid] += red[tid + o]; __syncthreads(); }
  if (tid == 0) {
    float m = stats[0];
    float var = red[0] * (1.f / 512.f) - m * m;
    float sd = sqrtf(var + 1e-5f);
    stats[1] = sd; meanw[bn] = m; stdw[bn] = sd;
  }
  __syncthreads();
  const float m = stats[0], inv = 1.f / stats[1];
  for (int t = tid; t < 512; t += 256) xn[t] = (xn[t] - m) * inv;
  // load conv weights transposed for broadcast-friendly reads
  for (int idx = tid; idx < 1536; idx += 256) {
    int o = idx / 48, r = idx - o * 48, i = r / 3, t = r - i * 3;
    cw2[(i * 3 + t) * 32 + o] = conv_w[idx];
  }
  __syncthreads();
  if (tid < 8) xn[512 + tid] = xn[511];   // replicate-pad last STRIDE samples
  __syncthreads();
  for (int idx = tid; idx < LPAT * DMODEL; idx += 256) {
    int l = idx >> 5, o = idx & 31;
    float a = 0.f;
    #pragma unroll
    for (int t = 0; t < 3; ++t) {
      int pi = (l + t + 63) & 63;          // circular pad on patch axis
      const float* xb = xn + 8 * pi;
      #pragma unroll
      for (int i = 0; i < 16; ++i) a += cw2[(i * 3 + t) * 32 + o] * xb[i];
    }
    enc[(size_t)bn * 2048 + idx] = a;
  }
}

// =====================================================================
// Kernel 3 (GEMM1, WMMA bf16): Et[h][v] = sum_d word_emb[v][d] * Bkv[h][d]
// M=32000 (v), N=512 (h), K=4096. Tile 64x128, 8 waves, KT=64.
// Only the HBM-resident A stream (word_emb) is register-pipelined; the
// L2-resident B operand is copied global->LDS with short-lived registers
// (avoids the scratch spills seen with dual-stream pipelining).
// Output stored transposed bf16 (packed 16B stores per lane).
// =====================================================================
__global__ __launch_bounds__(256)
void k_gemm1(const float* __restrict__ word_emb, const __bf16* __restrict__ Bkv,
             __bf16* __restrict__ Et) {
  __shared__ __bf16 As[64][72];
  __shared__ __bf16 Bs[128][72];
  const int tid = threadIdx.x, lane = tid & 31, wid = tid >> 5;
  const int row0 = blockIdx.x * 64;        // v
  const int col0 = blockIdx.y * 128;       // h
  const int wm = (wid >> 1) * 16;
  const int wn = (wid & 1) * 64;
  const int g = lane >> 4, q15 = lane & 15;
  v8f acc[4] = {};
  const int ar = tid >> 2, ac = (tid & 3) * 16;   // A stage: 16 f32 each
  const int br = tid >> 1, bc = (tid & 1) * 32;   // B stage: 32 bf16 each
  const float*  gA = word_emb + (size_t)(row0 + ar) * DLLM + ac;
  const __bf16* gB = Bkv + (size_t)(col0 + br) * DLLM + bc;

  float4 ra[4];
  #pragma unroll
  for (int q = 0; q < 4; ++q) ra[q] = *(const float4*)(gA + 4 * q);

  for (int k0 = 0; k0 < DLLM; k0 += 64) {
    // commit pipelined A registers to LDS
    *(bf16x8*)&As[ar][ac]     = cvt8(ra[0], ra[1]);
    *(bf16x8*)&As[ar][ac + 8] = cvt8(ra[2], ra[3]);
    // direct global->LDS copy of the L2-hot B tile (short live range)
    {
      uint4* dB = (uint4*)&Bs[br][bc];
      #pragma unroll
      for (int q = 0; q < 4; ++q) dB[q] = *(const uint4*)(gB + k0 + 8 * q);
    }
    __syncthreads();
    // issue next chunk's A loads early (latency hidden by WMMAs)
    if (k0 + 64 < DLLM) {
      #pragma unroll
      for (int q = 0; q < 4; ++q) ra[q] = *(const float4*)(gA + (k0 + 64) + 4 * q);
    }
    // fragments grouped per half-chunk, WMMAs issue back-to-back
    {
      v16bf a0 = ld_frag_a(&As[wm + q15][8 * g]);
      v16bf bf[4];
      #pragma unroll
      for (int t = 0; t < 4; ++t) bf[t] = ld_frag_b(&Bs[wn + 16 * t + q15][8 * g]);
      #pragma unroll
      for (int t = 0; t < 4; ++t) acc[t] = wmma_bf16(a0, bf[t], acc[t]);
    }
    {
      v16bf a1 = ld_frag_a(&As[wm + q15][32 + 8 * g]);
      v16bf bf[4];
      #pragma unroll
      for (int t = 0; t < 4; ++t) bf[t] = ld_frag_b(&Bs[wn + 16 * t + q15][32 + 8 * g]);
      #pragma unroll
      for (int t = 0; t < 4; ++t) acc[t] = wmma_bf16(a1, bf[t], acc[t]);
    }
    __syncthreads();
  }
  // epilogue: Et[h][v] bf16, one packed 16B store per tile per lane
  #pragma unroll
  for (int t = 0; t < 4; ++t) {
    int h = col0 + wn + 16 * t + q15;
    size_t vb = (size_t)row0 + wm + 8 * g;
    bf16x8 w;
    #pragma unroll
    for (int r = 0; r < 8; ++r) w[r] = (__bf16)acc[t][r];
    *(bf16x8*)(Et + (size_t)h * VSZ + vb) = w;
  }
}

// =====================================================================
// Kernel 4 (GEMM2, WMMA bf16): KV[n][h] = sum_v map_w[n][v]*Et[h][v]
//                                         + map_b[n]*skv[h] + bkvb[h]
// M=1000 (tiles of 16, masked), N=512, K=32000. 4 waves, tile 16x128.
// A (map_w, HBM) register-pipelined; B (Et, L2-resident) copied directly.
// =====================================================================
__global__ __launch_bounds__(128)
void k_gemm2(const float* __restrict__ map_w, const __bf16* __restrict__ Et,
             const float* __restrict__ map_b, const float* __restrict__ skv,
             const float* __restrict__ bkvb, float* __restrict__ KV) {
  __shared__ __bf16 As[16][72];
  __shared__ __bf16 Bs[128][72];
  const int tid = threadIdx.x, lane = tid & 31, wid = tid >> 5;
  const int m0 = blockIdx.x * 16;          // token rows
  const int col0 = blockIdx.y * 128;       // h
  const int wn = wid * 32;
  const int g = lane >> 4, q15 = lane & 15;
  v8f acc[2] = {};
  const int ar = tid >> 3, ac = (tid & 7) * 8;   // A stage: 8 f32 each
  const bool arow_ok = (m0 + ar) < NTOK;
  const float*  gA = map_w + (size_t)(m0 + ar) * VSZ + ac;
  const __bf16* gB = Et + (size_t)(col0 + tid) * VSZ;

  float4 ra[2] = {make_float4(0.f, 0.f, 0.f, 0.f), make_float4(0.f, 0.f, 0.f, 0.f)};
  if (arow_ok) { ra[0] = *(const float4*)(gA); ra[1] = *(const float4*)(gA + 4); }

  for (int k0 = 0; k0 < VSZ; k0 += 64) {
    *(bf16x8*)&As[ar][ac] = cvt8(ra[0], ra[1]);
    {
      uint4* dB = (uint4*)&Bs[tid][0];
      #pragma unroll
      for (int q = 0; q < 8; ++q) dB[q] = *(const uint4*)(gB + k0 + 8 * q);
    }
    __syncthreads();
    if (k0 + 64 < VSZ && arow_ok) {
      ra[0] = *(const float4*)(gA + (k0 + 64));
      ra[1] = *(const float4*)(gA + (k0 + 64) + 4);
    }
    {
      v16bf a0 = ld_frag_a(&As[q15][8 * g]);
      v16bf bf[2];
      #pragma unroll
      for (int t = 0; t < 2; ++t) bf[t] = ld_frag_b(&Bs[wn + 16 * t + q15][8 * g]);
      #pragma unroll
      for (int t = 0; t < 2; ++t) acc[t] = wmma_bf16(a0, bf[t], acc[t]);
    }
    {
      v16bf a1 = ld_frag_a(&As[q15][32 + 8 * g]);
      v16bf bf[2];
      #pragma unroll
      for (int t = 0; t < 2; ++t) bf[t] = ld_frag_b(&Bs[wn + 16 * t + q15][32 + 8 * g]);
      #pragma unroll
      for (int t = 0; t < 2; ++t) acc[t] = wmma_bf16(a1, bf[t], acc[t]);
    }
    __syncthreads();
  }
  #pragma unroll
  for (int t = 0; t < 2; ++t) {
    int h = col0 + wn + 16 * t + q15;
    float sh = skv[h], bh = bkvb[h];
    #pragma unroll
    for (int r = 0; r < 8; ++r) {
      int m = m0 + r + 8 * g;
      if (m < NTOK) KV[(size_t)m * NKV + h] = acc[t][r] + map_b[m] * sh + bh;
    }
  }
}

// =====================================================================
// Kernel 5: fused Q-projection + flash attention per (bn, head).
// rep[bn][l][h*32+e] = softmax(QK^T/sqrt(32)) V
// =====================================================================
__global__ __launch_bounds__(256)
void k_attn(const float* __restrict__ enc, const float* __restrict__ KV,
            const float* __restrict__ wq, const float* __restrict__ bq,
            float* __restrict__ rep) {
  __shared__ float encs[64][33];
  __shared__ float Qs[64][33];
  __shared__ float Ks[64][33];
  __shared__ float Vs[64][33];
  __shared__ float wqs[32][33];
  __shared__ float Ps[8][8][64];
  const int bn = blockIdx.x, h = blockIdx.y;
  const int tid = threadIdx.x, lane = tid & 31, wave = tid >> 5;
  const float NEG = -1e30f;
  const float scale = 0.17677669529663687f;   // 1/sqrt(32)

  for (int i = 0; i < 8; ++i) {
    int idx = tid + 256 * i;
    encs[idx >> 5][idx & 31] = enc[(size_t)bn * 2048 + idx];
  }
  for (int i = 0; i < 4; ++i) {
    int idx = tid + 256 * i;
    wqs[idx >> 5][idx & 31] = wq[(size_t)(h * 32 + (idx >> 5)) * 32 + (idx & 31)];
  }
  __syncthreads();
  for (int i = 0; i < 8; ++i) {
    int idx = tid + 256 * i;
    int l = idx >> 5, e = idx & 31;
    float a = bq[h * 32 + e];
    #pragma unroll
    for (int mm = 0; mm < 32; ++mm) a += encs[l][mm] * wqs[e][mm];
    Qs[l][e] = a * scale;
  }
  __syncthreads();

  float accv[8], mrow[8], drow[8], alph[8];
  #pragma unroll
  for (int r = 0; r < 8; ++r) { accv[r] = 0.f; mrow[r] = NEG; drow[r] = 0.f; }

  for (int tile = 0; tile < 16; ++tile) {
    const int s0 = tile * 64;
    for (int i = 0; i < 8; ++i) {
      int idx = tid + 256 * i;
      int row = idx >> 5, e = idx & 31;
      int srow = s0 + row;
      bool v = srow < NTOK;
      Ks[row][e] = v ? KV[(size_t)srow * NKV + h * 32 + e] : 0.f;
      Vs[row][e] = v ? KV[(size_t)srow * NKV + HF + h * 32 + e] : 0.f;
    }
    __syncthreads();
    #pragma unroll
    for (int r = 0; r < 8; ++r) {
      const int l = wave * 8 + r;
      const int sa = s0 + lane, sb = s0 + lane + 32;
      float sc0 = 0.f, sc1 = 0.f;
      #pragma unroll
      for (int e = 0; e < 32; ++e) {
        float qv = Qs[l][e];
        sc0 += qv * Ks[lane][e];
        sc1 += qv * Ks[lane + 32][e];
      }
      if (sa >= NTOK) sc0 = NEG;
      if (sb >= NTOK) sc1 = NEG;
      float tmax = fmaxf(sc0, sc1);
      #pragma unroll
      for (int d = 16; d > 0; d >>= 1) tmax = fmaxf(tmax, __shfl_xor(tmax, d, 32));
      float mnew = fmaxf(mrow[r], tmax);
      float p0 = (sa < NTOK) ? __expf(sc0 - mnew) : 0.f;
      float p1 = (sb < NTOK) ? __expf(sc1 - mnew) : 0.f;
      float ps = p0 + p1;
      #pragma unroll
      for (int d = 16; d > 0; d >>= 1) ps += __shfl_xor(ps, d, 32);
      alph[r] = __expf(mrow[r] - mnew);
      drow[r] = drow[r] * alph[r] + ps;
      mrow[r] = mnew;
      Ps[wave][r][lane] = p0;
      Ps[wave][r][lane + 32] = p1;
    }
    __syncthreads();
    #pragma unroll
    for (int r = 0; r < 8; ++r) {
      float a = 0.f;
      #pragma unroll 8
      for (int s = 0; s < 64; ++s) a += Ps[wave][r][s] * Vs[s][lane];
      accv[r] = accv[r] * alph[r] + a;
    }
    __syncthreads();
  }
  #pragma unroll
  for (int r = 0; r < 8; ++r) {
    int l = wave * 8 + r;
    rep[((size_t)bn * 64 + l) * HF + h * 32 + lane] = accv[r] / drow[r];
  }
}

// =====================================================================
// Kernel 6: fused out-proj (only first 32 cols of wo!), head, de-norm.
// =====================================================================
__global__ __launch_bounds__(256)
void k_head(const float* __restrict__ rep, const float* __restrict__ wo,
            const float* __restrict__ bo, const float* __restrict__ head_w,
            const float* __restrict__ head_b, const float* __restrict__ meanw,
            const float* __restrict__ stdw, float* __restrict__ out) {
  __shared__ float repS[64 * 256];
  __shared__ float o32[64][33];
  const int bn = blockIdx.x, b = bn / 7, n = bn - b * 7;
  const int tid = threadIdx.x;

  const float4* g = (const float4*)(rep + (size_t)bn * 16384);
  float4* sp = (float4*)repS;
  for (int i = 0; i < 16; ++i) sp[tid + 256 * i] = g[tid + 256 * i];
  __syncthreads();
  for (int i = 0; i < 8; ++i) {
    int idx = tid + 256 * i;
    int l = idx >> 5, e = idx & 31;
    float a = bo[e];
    const float* rrow = repS + l * 256;
    const float* wrow = wo + (size_t)e * HF;
    #pragma unroll 8
    for (int j = 0; j < 256; ++j) a += rrow[j] * wrow[j];
    o32[l][e] = a;
  }
  __syncthreads();
  if (tid < 96) {
    const int p = tid;
    float a = head_b[p];
    const float* hw = head_w + (size_t)p * 2048;
    for (int e = 0; e < 32; ++e)
      #pragma unroll 8
      for (int l = 0; l < 64; ++l) a += o32[l][e] * hw[e * 64 + l];
    out[((size_t)b * 96 + p) * 7 + n] = a * stdw[bn] + meanw[bn];
  }
}

// =====================================================================
extern "C" void kernel_launch(void* const* d_in, const int* in_sizes, int n_in,
                              void* d_out, int out_size, void* d_ws, size_t ws_size,
                              hipStream_t stream) {
  (void)in_sizes; (void)n_in; (void)out_size; (void)ws_size;
  const float* x_enc    = (const float*)d_in[0];
  const float* conv_w   = (const float*)d_in[1];
  const float* word_emb = (const float*)d_in[2];
  const float* map_w    = (const float*)d_in[3];
  const float* map_b    = (const float*)d_in[4];
  const float* wq       = (const float*)d_in[5];
  const float* bq       = (const float*)d_in[6];
  const float* wk       = (const float*)d_in[7];
  const float* bk       = (const float*)d_in[8];
  const float* wv       = (const float*)d_in[9];
  const float* bv       = (const float*)d_in[10];
  const float* wo       = (const float*)d_in[11];
  const float* bo       = (const float*)d_in[12];
  const float* head_w   = (const float*)d_in[13];
  const float* head_b   = (const float*)d_in[14];
  float* out = (float*)d_out;
  char* ws = (char*)d_ws;

  // workspace layout (~60 MB total)
  float*  meanw = (float*)(ws + 0);                    // 224 f32
  float*  stdw  = (float*)(ws + 1024);                 // 224 f32
  float*  skv   = (float*)(ws + 2048);                 // 512 f32
  float*  bkvb  = (float*)(ws + 4096);                 // 512 f32
  float*  enc   = (float*)(ws + 8192);                 // 224*64*32 f32 (1.75 MB)
  __bf16* Bkv   = (__bf16*)(ws + (size_t)(2  << 20));  // 512*4096 bf16 (4 MB)
  __bf16* Et    = (__bf16*)(ws + (size_t)(8  << 20));  // 512*32000 bf16 (31.3 MB)
  float*  KV    = (float*) (ws + (size_t)(42 << 20));  // 1000*512 f32 (2 MB)
  float*  rep   = (float*) (ws + (size_t)(45 << 20));  // 224*64*256 f32 (14 MB)

  k_prep_bkv<<<512, 256, 0, stream>>>(wk, wv, bk, bv, Bkv, skv, bkvb);
  k_stats_enc<<<BN_CNT, 256, 0, stream>>>(x_enc, conv_w, enc, meanw, stdw);
  k_gemm1<<<dim3(VSZ / 64, NKV / 128), 256, 0, stream>>>(word_emb, Bkv, Et);
  k_gemm2<<<dim3(63, NKV / 128), 128, 0, stream>>>(map_w, Et, map_b, skv, bkvb, KV);
  k_attn<<<dim3(BN_CNT, 8), 256, 0, stream>>>(enc, KV, wq, bq, rep);
  k_head<<<BN_CNT, 256, 0, stream>>>(rep, wo, bo, head_w, head_b, meanw, stdw, out);
}
